// MultiheadAttention_33904471835012
// MI455X (gfx1250) — matrix operand order repaired and checked by reading, not compiled, not run on hardware
//
#include <hip/hip_runtime.h>

// MI455X (gfx1250): ~34 GFLOP total, bf16 working set ~20MB << 192MB L2.
// Pipeline: one-time f32->bf16 conversion (activations row-major, weights
// transposed), TDM-fed double-buffered WMMA GEMMs, score recompute from L2
// for the column-softmax (over i) attention, all via v_wmma_f32_16x16x32_bf16.

typedef __bf16 bf16;
typedef __attribute__((ext_vector_type(16))) __bf16 v16bf;
typedef __attribute__((ext_vector_type(8)))  float  v8f;
typedef __attribute__((ext_vector_type(4)))  unsigned int v4u;
typedef __attribute__((ext_vector_type(8)))  int v8i;
typedef __attribute__((ext_vector_type(4)))  int v4i;

#define I_LEN 2048
#define J_LEN 2048
#define MDIM  1024
#define HNUM  64      // einsum head axis (weight dim1)
#define KPH   16      // contracted per-head feature (weight dim2)

#if __has_builtin(__builtin_amdgcn_tensor_load_to_lds)
#define HAVE_TDM 1
#else
#define HAVE_TDM 0
#endif

__device__ inline void store_c(float* p, float v) { *p = v; }
__device__ inline void store_c(bf16* p,  float v) { *p = (bf16)v; }

#if HAVE_TDM
// Issue a TDM load of a 128-row x 32-col bf16 tile (k-contiguous rows,
// row stride = rowStride elements) into LDS at byte offset ldsOff.
// D# layout per CDNA5 ISA.
// This toolchain's builtin takes 6 args: (v4u, v8i, v4i, v4i, v8i, i32 cpol);
// the trailing groups are unused for a 2-D tile (tile_dim2 = 0).
__device__ inline void tdm_load_tile_128x32(unsigned int ldsOff,
                                            const bf16* gptr,
                                            unsigned int rowStride) {
    const unsigned long long ga = (unsigned long long)(uintptr_t)gptr;
    v4u g0;
    g0[0] = 1u;                                   // count=1, user mode, no gather
    g0[1] = ldsOff;                               // lds_addr (bytes)
    g0[2] = (unsigned int)ga;                     // global_addr[31:0]
    g0[3] = (unsigned int)(ga >> 32) | (2u << 30);// addr[56:32] | type=2 (image)
    v8i g1;
    const unsigned int td0 = 1u << 20;            // generous in-bounds tensor dims
    const unsigned int td1 = 1u << 20;
    g1[0] = (int)(1u << 16);                      // data_size=1 (2 bytes)
    g1[1] = (int)((td0 & 0xFFFFu) << 16);         // tensor_dim0[15:0]
    g1[2] = (int)((td0 >> 16) | ((td1 & 0xFFFFu) << 16));
    g1[3] = (int)((td1 >> 16) | (32u << 16));     // tile_dim0 = 32 (k)
    g1[4] = (int)128u;                            // tile_dim1 = 128 rows, dim2=0
    g1[5] = (int)rowStride;                       // tensor_dim0_stride[31:0]
    g1[6] = 0;                                    // stride[47:32]=0, dim1_stride lo=0
    g1[7] = 0;
    const v4i gz4 = {0, 0, 0, 0};                 // groups 2/3 unused (2-D tile)
    const v8i gz8 = {0, 0, 0, 0, 0, 0, 0, 0};
    __builtin_amdgcn_tensor_load_to_lds(g0, g1, gz4, gz4, gz8, 0);
}
#endif

// ---------------------------------------------------------------------------
// One-time conversions: activations f32 -> bf16 (row-major),
// weights f32 [K][N] -> bf16 transposed Wt[N][K].
// ---------------------------------------------------------------------------
__global__ __launch_bounds__(256) void conv_bf16(const float* __restrict__ in,
                                                 bf16* __restrict__ out) {
    const int i = (blockIdx.x * 256 + threadIdx.x) * 4;
    const float4 v = *(const float4*)(in + i);
    out[i + 0] = (bf16)v.x;
    out[i + 1] = (bf16)v.y;
    out[i + 2] = (bf16)v.z;
    out[i + 3] = (bf16)v.w;
}

__global__ __launch_bounds__(256) void transpose_bf16(const float* __restrict__ W,
                                                      bf16* __restrict__ Wt) {
    __shared__ bf16 tile[32][33];
    const int k0 = blockIdx.y * 32, n0 = blockIdx.x * 32;
    const int r = threadIdx.x >> 5, c = threadIdx.x & 31;
#pragma unroll
    for (int rr = r; rr < 32; rr += 8)
        tile[rr][c] = (bf16)W[(size_t)(k0 + rr) * MDIM + n0 + c];
    __syncthreads();
#pragma unroll
    for (int rr = r; rr < 32; rr += 8)
        Wt[(size_t)(n0 + rr) * MDIM + k0 + c] = tile[c][rr];
}

// ---------------------------------------------------------------------------
// 128x128-tile GEMM: C[M,N] = A[M,K](bf16) * Wt[N,K](bf16)^T.
// 8 waves (2x4), each 64x32 via 4x2 WMMA tiles, K-step 32.
// Tiles are DMA'd into LDS by the Tensor Data Mover, double buffered:
// stage s+1 is in flight while stage s computes (s_wait_tensorcnt 2).
// ---------------------------------------------------------------------------
template <typename CT>
__global__ __launch_bounds__(256) void gemm_wmma(const bf16* __restrict__ A,
                                                 const bf16* __restrict__ Wt,
                                                 CT* __restrict__ C,
                                                 int Mdim, int Ndim, int Kdim) {
    __shared__ bf16 As[2][128][32];   // [m][k]
    __shared__ bf16 Bs[2][128][32];   // [n][k]

    const int bn   = blockIdx.x * 128;
    const int bm   = blockIdx.y * 128;
    const int t    = threadIdx.x;
    const int wave = t >> 5;
    const int lane = t & 31;
    const int half = lane >> 4;
    const int ln   = lane & 15;
    const int wm   = (wave & 1) * 64;
    const int wn   = (wave >> 1) * 32;

    v8f acc[4][2];
#pragma unroll
    for (int i = 0; i < 4; ++i)
#pragma unroll
        for (int j = 0; j < 2; ++j) acc[i][j] = (v8f){0,0,0,0,0,0,0,0};

    const int nsteps = Kdim >> 5;

#if HAVE_TDM
    unsigned int aOff[2], bOff[2];
#pragma unroll
    for (int b = 0; b < 2; ++b) {
        aOff[b] = (unsigned int)(uintptr_t)&As[b][0][0];
        bOff[b] = (unsigned int)(uintptr_t)&Bs[b][0][0];
    }
    if (wave == 0) {
        tdm_load_tile_128x32(aOff[0], A  + (size_t)bm * Kdim, (unsigned)Kdim);
        tdm_load_tile_128x32(bOff[0], Wt + (size_t)bn * Kdim, (unsigned)Kdim);
    }
#endif

    for (int s = 0; s < nsteps; ++s) {
        const int buf = s & 1;
#if HAVE_TDM
        if (wave == 0) {
            if (s + 1 < nsteps) {
                const int kk = (s + 1) << 5;
                tdm_load_tile_128x32(aOff[buf ^ 1], A  + (size_t)bm * Kdim + kk, (unsigned)Kdim);
                tdm_load_tile_128x32(bOff[buf ^ 1], Wt + (size_t)bn * Kdim + kk, (unsigned)Kdim);
                __builtin_amdgcn_s_wait_tensorcnt(2);  // stage s landed (in-order)
            } else {
                __builtin_amdgcn_s_wait_tensorcnt(0);
            }
        }
        __syncthreads();
#else
        __syncthreads();
        {
            const int kk   = s << 5;
            const int row  = t >> 1;
            const int chal = (t & 1) * 16;
            const bf16* sa = A  + (size_t)(bm + row) * Kdim + kk + chal;
            const bf16* sb = Wt + (size_t)(bn + row) * Kdim + kk + chal;
#pragma unroll
            for (int e = 0; e < 16; ++e) As[buf][row][chal + e] = sa[e];
#pragma unroll
            for (int e = 0; e < 16; ++e) Bs[buf][row][chal + e] = sb[e];
        }
        __syncthreads();
#endif
        v16bf afr[4];
#pragma unroll
        for (int ms = 0; ms < 4; ++ms) {
            const int r = wm + ms * 16 + ln;
#pragma unroll
            for (int e = 0; e < 16; ++e) {
                const int k = half * 16 + e; // contiguous k = 16*half + e
                afr[ms][e] = As[buf][r][k];
            }
        }
#pragma unroll
        for (int ns = 0; ns < 2; ++ns) {
            const int n = wn + ns * 16 + ln;
            v16bf bfr;
#pragma unroll
            for (int e = 0; e < 16; ++e) bfr[e] = Bs[buf][n][half * 16 + e]; // contiguous k = 16*half + e, same as A
#pragma unroll
            for (int ms = 0; ms < 4; ++ms)
                acc[ms][ns] = __builtin_amdgcn_wmma_f32_16x16x32_bf16(
                    false, afr[ms], false, bfr, (short)0, acc[ms][ns], false, false);
        }
#if HAVE_TDM
        __syncthreads();   // all waves done with buf before it is re-filled
#endif
    }

#pragma unroll
    for (int ms = 0; ms < 4; ++ms)
#pragma unroll
        for (int ns = 0; ns < 2; ++ns)
#pragma unroll
            for (int e = 0; e < 8; ++e) {
                const int m = bm + wm + ms * 16 + e + 8 * half; // ISA C/D layout
                const int n = bn + wn + ns * 16 + ln;
                store_c(&C[(size_t)m * Ndim + n], acc[ms][ns][e]);
            }
}

// ---------------------------------------------------------------------------
// Pass 1: column-wise (over i) softmax stats per (head, j-strip of 16).
// ---------------------------------------------------------------------------
__global__ __launch_bounds__(256) void attn_stats(const bf16* __restrict__ WQ,
                                                  const bf16* __restrict__ WK,
                                                  float* __restrict__ Mh,
                                                  float* __restrict__ Dh) {
    const int wave = threadIdx.x >> 5;
    const int lane = threadIdx.x & 31;
    const int half = lane >> 4;
    const int ln   = lane & 15;

    const int strip = blockIdx.x * 8 + wave;   // 0..8191
    const int h  = strip >> 7;                 // 0..63
    const int j0 = (strip & 127) << 4;         // 0..2032

    v16bf b = {};                              // B[k][n]=WK[j0+n][h*16+k], k<16
    if (half == 0) {
        const bf16* p = WK + (size_t)(j0 + ln) * MDIM + h * KPH;
#pragma unroll
        for (int e = 0; e < 16; ++e) b[e] = p[e];
    }

    float rmax = -1.0e30f, rsum = 0.0f;
    for (int i0 = 0; i0 < I_LEN; i0 += 16) {
        v16bf a = {};                          // A[r][k]=WQ[i0+r][h*16+k], k<16
        {
            const bf16* p = WQ + (size_t)(i0 + ln) * MDIM + h * KPH;
            if (half == 0)
#pragma unroll
            for (int e = 0; e < 16; ++e) a[e] = p[e];
        }
        v8f c = (v8f){0,0,0,0,0,0,0,0};
        c = __builtin_amdgcn_wmma_f32_16x16x32_bf16(false, a, false, b,
                                                    (short)0, c, false, false);
        float tmax = -1.0e30f;
#pragma unroll
        for (int e = 0; e < 8; ++e) {
            const float s = c[e] * 0.125f;     // / sqrt(64)
            c[e] = s;
            tmax = fmaxf(tmax, s);
        }
        tmax = fmaxf(tmax, __shfl_xor(tmax, 16, 32));
        const float nmax = fmaxf(rmax, tmax);
        float tsum = 0.0f;
#pragma unroll
        for (int e = 0; e < 8; ++e) tsum += __expf(c[e] - nmax);
        tsum += __shfl_xor(tsum, 16, 32);
        rsum = rsum * __expf(rmax - nmax) + tsum;
        rmax = nmax;
    }
    if (half == 0) {
        Mh[h * J_LEN + j0 + ln] = rmax;
        Dh[h * J_LEN + j0 + ln] = rsum;
    }
}

// ---------------------------------------------------------------------------
// Pass 2: heads[i, h*16+v] = sum_j exp(s-M[h,j])/D[h,j] * WV[j][h*16+v].
// ---------------------------------------------------------------------------
__global__ __launch_bounds__(256) void attn_apply(const bf16* __restrict__ WQ,
                                                  const bf16* __restrict__ WK,
                                                  const bf16* __restrict__ WV,
                                                  const float* __restrict__ Mh,
                                                  const float* __restrict__ Dh,
                                                  bf16* __restrict__ heads) {
    __shared__ bf16 wvt[16][512];      // WVt chunk: [v][j-local]
    __shared__ bf16 pbuf[8][16][32];   // per-wave P tile (i x j)

    const int h    = blockIdx.y;
    const int wave = threadIdx.x >> 5;
    const int i0   = blockIdx.x * 128 + wave * 16;
    const int lane = threadIdx.x & 31;
    const int half = lane >> 4;
    const int ln   = lane & 15;

    v16bf aq = {};                     // loop-invariant WQ fragment (k>=16 zero)
    {
        const bf16* p = WQ + (size_t)(i0 + ln) * MDIM + h * KPH;
        if (half == 0)
#pragma unroll
        for (int e = 0; e < 16; ++e) aq[e] = p[e];
    }

    const float* Mrow = Mh + h * J_LEN;
    const float* Drow = Dh + h * J_LEN;
    bf16* pb = &pbuf[wave][0][0];
    v8f acc = (v8f){0,0,0,0,0,0,0,0};

    for (int jc = 0; jc < J_LEN; jc += 512) {
        __syncthreads();
        for (int j = threadIdx.x; j < 512; j += 256) {
            const bf16* src = WV + (size_t)(jc + j) * MDIM + h * KPH;
#pragma unroll
            for (int v = 0; v < 16; ++v) wvt[v][j] = src[v];
        }
        __syncthreads();

        for (int j0 = 0; j0 < 512; j0 += 32) {
            const int jg = jc + j0;
            v8f s0 = (v8f){0,0,0,0,0,0,0,0};
            v8f s1 = (v8f){0,0,0,0,0,0,0,0};
            {
                v16bf b = {};
                if (half == 0) {
                    const bf16* p = WK + (size_t)(jg + ln) * MDIM + h * KPH;
#pragma unroll
                    for (int e = 0; e < 16; ++e) b[e] = p[e];
                }
                s0 = __builtin_amdgcn_wmma_f32_16x16x32_bf16(false, aq, false, b,
                                                             (short)0, s0, false, false);
            }
            {
                v16bf b = {};
                if (half == 0) {
                    const bf16* p = WK + (size_t)(jg + 16 + ln) * MDIM + h * KPH;
#pragma unroll
                    for (int e = 0; e < 16; ++e) b[e] = p[e];
                }
                s1 = __builtin_amdgcn_wmma_f32_16x16x32_bf16(false, aq, false, b,
                                                             (short)0, s1, false, false);
            }
            const float m0  = Mrow[jg + ln];
            const float id0 = 1.0f / Drow[jg + ln];
            const float m1  = Mrow[jg + 16 + ln];
            const float id1 = 1.0f / Drow[jg + 16 + ln];
#pragma unroll
            for (int e = 0; e < 8; ++e) {
                const int m = half * 8 + e;
                pb[m * 32 + ln]      = (bf16)(__expf(s0[e] * 0.125f - m0) * id0);
                pb[m * 32 + 16 + ln] = (bf16)(__expf(s1[e] * 0.125f - m1) * id1);
            }
            asm volatile("s_wait_dscnt 0" ::: "memory");
            v16bf ap, bv;
#pragma unroll
            for (int e = 0; e < 16; ++e) {
                const int k = half * 16 + e;
                ap[e] = pb[ln * 32 + k];
            }
#pragma unroll
            for (int e = 0; e < 16; ++e) bv[e] = wvt[ln][j0 + half * 16 + e];
            acc = __builtin_amdgcn_wmma_f32_16x16x32_bf16(false, ap, false, bv,
                                                          (short)0, acc, false, false);
        }
    }
#pragma unroll
    for (int e = 0; e < 8; ++e) {
        const int m = e + 8 * half;
        heads[(size_t)(i0 + m) * MDIM + h * KPH + ln] = (bf16)acc[e];
    }
}

// ---------------------------------------------------------------------------
extern "C" void kernel_launch(void* const* d_in, const int* in_sizes, int n_in,
                              void* d_out, int out_size, void* d_ws, size_t ws_size,
                              hipStream_t stream) {
    (void)in_sizes; (void)n_in; (void)out_size; (void)ws_size;
    const float* query = (const float*)d_in[0];
    const float* key   = (const float*)d_in[1];
    const float* value = (const float*)d_in[2];
    const float* q_w   = (const float*)d_in[3];
    const float* k_w   = (const float*)d_in[4];
    const float* v_w   = (const float*)d_in[5];
    const float* o_w   = (const float*)d_in[6];
    float* out = (float*)d_out;

    char* ws = (char*)d_ws;
    bf16*  WQ  = (bf16*)(ws);                        // 4 MiB
    bf16*  WK  = (bf16*)(ws + (4u  << 20));          // 4 MiB
    bf16*  WV  = (bf16*)(ws + (8u  << 20));          // 4 MiB
    bf16*  HB  = (bf16*)(ws + (12u << 20));          // 4 MiB
    float* Mh  = (float*)(ws + (16u << 20));         // 512 KiB
    float* Dh  = (float*)(ws + (16u << 20) + (512u << 10));
    bf16*  Qb  = (bf16*)(ws + (17u << 20));          // 4 MiB
    bf16*  Kb  = (bf16*)(ws + (21u << 20));          // 4 MiB
    bf16*  Vb  = (bf16*)(ws + (25u << 20));          // 4 MiB
    bf16*  qWt = (bf16*)(ws + (29u << 20));          // 2 MiB each
    bf16*  kWt = (bf16*)(ws + (31u << 20));
    bf16*  vWt = (bf16*)(ws + (33u << 20));
    bf16*  oWt = (bf16*)(ws + (35u << 20));

    const dim3 blk(256);

    // one-time conversions
    const int nAct = (I_LEN * MDIM) / 1024;          // 4 elems/thread
    conv_bf16<<<dim3(nAct), blk, 0, stream>>>(query, Qb);
    conv_bf16<<<dim3(nAct), blk, 0, stream>>>(key,   Kb);
    conv_bf16<<<dim3(nAct), blk, 0, stream>>>(value, Vb);
    const dim3 gt(MDIM / 32, MDIM / 32);
    transpose_bf16<<<gt, blk, 0, stream>>>(q_w, qWt);
    transpose_bf16<<<gt, blk, 0, stream>>>(k_w, kWt);
    transpose_bf16<<<gt, blk, 0, stream>>>(v_w, vWt);
    transpose_bf16<<<gt, blk, 0, stream>>>(o_w, oWt);

    // projections (TDM-fed WMMA GEMMs), bf16 out
    const dim3 gg(MDIM / 128, I_LEN / 128);          // (8, 16)
    gemm_wmma<bf16><<<gg, blk, 0, stream>>>(Qb, qWt, WQ, I_LEN, MDIM, MDIM);
    gemm_wmma<bf16><<<gg, blk, 0, stream>>>(Kb, kWt, WK, J_LEN, MDIM, MDIM);
    gemm_wmma<bf16><<<gg, blk, 0, stream>>>(Vb, vWt, WV, J_LEN, MDIM, MDIM);

    // column softmax stats over i
    attn_stats<<<dim3(1024), blk, 0, stream>>>(WQ, WK, Mh, Dh);

    // weighted sum over j -> heads (bf16)
    attn_apply<<<dim3(I_LEN / 128, HNUM), blk, 0, stream>>>(WQ, WK, WV, Mh, Dh, HB);

    // output projection, fp32 result
    gemm_wmma<float><<<gg, blk, 0, stream>>>(HB, oWt, out, I_LEN, MDIM, MDIM);
}
